// Block_38860864094289
// MI455X (gfx1250) — matrix operand order repaired and checked by reading, not compiled, not run on hardware
//
#include <hip/hip_runtime.h>
#include <hip/hip_bf16.h>
#include <math.h>

// ---------------------------------------------------------------------------
// Types for CDNA5 WMMA (wave32, v_wmma_f32_16x16x32_bf16)
// ---------------------------------------------------------------------------
typedef __bf16 bf16_t;
typedef __attribute__((ext_vector_type(8)))  __bf16 v8bf;
typedef __attribute__((ext_vector_type(16))) __bf16 v16bf;
typedef __attribute__((ext_vector_type(8)))  float  v8f;

#define BN 32
#define FN 81
#define CN 512
#define HIDN 2048
#define HALF_F 40           // rows in branch 1
#define F2 41               // rows in branch 2
#define HH 21               // dwt height (41 -> pad 42 -> 21)
#define HW 256              // dwt width  (512 -> 256)
#define INV_SQRT2f 0.70710678118654752f

static __device__ __forceinline__ bf16_t f2bf(float f) {
    unsigned u = __builtin_bit_cast(unsigned, f);
    unsigned r = (u + 0x7fffu + ((u >> 16) & 1u)) >> 16;
    return __builtin_bit_cast(bf16_t, (unsigned short)r);
}
static __device__ __forceinline__ float gelu_exact(float x) {
    return 0.5f * x * (1.0f + erff(x * INV_SQRT2f));
}

// ---------------------------------------------------------------------------
// Weight prep: fp32 [K,N] row-major  ->  bf16 [N,K] row-major (B^T)
// ---------------------------------------------------------------------------
__global__ void k_transpose_bf16(const float* __restrict__ w, bf16_t* __restrict__ wT,
                                 int K, int N) {
    int idx = blockIdx.x * blockDim.x + threadIdx.x;
    if (idx >= K * N) return;
    int k = idx / N, n = idx % N;
    wT[(size_t)n * K + k] = f2bf(w[idx]);
}

// ---------------------------------------------------------------------------
// LayerNorm over rows of length 512; optional fp32 and bf16 outputs
// ---------------------------------------------------------------------------
__global__ __launch_bounds__(256) void k_ln512(const float* __restrict__ in,
                                               const float* __restrict__ w,
                                               const float* __restrict__ b,
                                               float* __restrict__ outF,
                                               bf16_t* __restrict__ outH) {
    __shared__ float s1[256], s2[256];
    int row = blockIdx.x, tid = threadIdx.x;
    const float* xr = in + (size_t)row * CN;
    float x0 = xr[tid], x1 = xr[tid + 256];
    s1[tid] = x0 + x1;
    s2[tid] = x0 * x0 + x1 * x1;
    for (int o = 128; o > 0; o >>= 1) {
        __syncthreads();
        if (tid < o) { s1[tid] += s1[tid + o]; s2[tid] += s2[tid + o]; }
    }
    __syncthreads();
    float mean = s1[0] * (1.0f / CN);
    float var  = s2[0] * (1.0f / CN) - mean * mean;
    float rstd = rsqrtf(var + 1e-5f);
    float y0 = (x0 - mean) * rstd * w[tid]       + b[tid];
    float y1 = (x1 - mean) * rstd * w[tid + 256] + b[tid + 256];
    size_t o0 = (size_t)row * CN + tid, o1 = o0 + 256;
    if (outF) { outF[o0] = y0; outF[o1] = y1; }
    if (outH) { outH[o0] = f2bf(y0); outH[o1] = f2bf(y1); }
}

// ---------------------------------------------------------------------------
// WMMA bf16 GEMM: out[M,N] = act(A[M,K] @ B + bias) + res
//   A   : bf16 row-major [M,K]
//   BT  : bf16 row-major [N,K]   (pre-transposed B)
//   8 waves/block, 2(M) x 4(N); each wave owns a 16x64 strip (4 accumulators)
//   -> A fragment loaded once per k-step and reused across 4 WMMAs.
//   Requires M % 32 == 0 handled by guard; N % 256 == 0 for full blocks.
// ---------------------------------------------------------------------------
__global__ __launch_bounds__(256) void k_gemm_wmma(const bf16_t* __restrict__ A,
                                                   const bf16_t* __restrict__ BT,
                                                   const float* __restrict__ bias,
                                                   const float* __restrict__ res,
                                                   float* __restrict__ outF,
                                                   bf16_t* __restrict__ outH,
                                                   int M, int N, int K, int act_gelu) {
    int lane = threadIdx.x & 31;
    int wave = threadIdx.x >> 5;
    int tm = blockIdx.x * 32 + (wave & 1) * 16;          // 2 waves in M
    int tn = blockIdx.y * 256 + (wave >> 1) * 64;        // 4 waves in N, 64 wide
    if (tm >= M || tn >= N) return;                      // wave-uniform guard
    int half = lane >> 4, l16 = lane & 15;

    const bf16_t* arow = A + (size_t)(tm + l16) * K;     // A frag: row m
    const bf16_t* brow[4];
#pragma unroll
    for (int t = 0; t < 4; ++t)                          // B frag: col n (4 tiles)
        brow[t] = BT + (size_t)(tn + 16 * t + l16) * K + 16 * half;

    v8f acc[4] = {};
    for (int kb = 0; kb < K; kb += 32) {
        // A 16x32 frag: elems 0..15 -> K = kb+16*half+0..15 (contiguous)
        v16bf af = *(const v16bf*)(arow + kb + 16 * half);
#pragma unroll
        for (int t = 0; t < 4; ++t) {
            // B 32x16 frag: elems 0..15 -> K = kb+16*half+0..15 (contiguous)
            v16bf bfrag = *(const v16bf*)(brow[t] + kb);
            acc[t] = __builtin_amdgcn_wmma_f32_16x16x32_bf16(
                         false, af, false, bfrag, (short)0, acc[t], false, false);
        }
    }

#pragma unroll
    for (int t = 0; t < 4; ++t) {
        int n = tn + 16 * t + l16;
        float bv = bias ? bias[n] : 0.0f;
#pragma unroll
        for (int r = 0; r < 8; ++r) {
            int m = tm + r + 8 * half;          // C/D layout: VGPR r -> M=r(+8)
            float v = acc[t][r] + bv;
            if (act_gelu) v = gelu_exact(v);
            if (res)  v += res[(size_t)m * N + n];
            size_t o = (size_t)m * N + n;
            if (outF) outF[o] = v;
            if (outH) outH[o] = f2bf(v);
        }
    }
}

// ---------------------------------------------------------------------------
// Additive attention: score[i,j]=sum_c relu(q_i+k_j); softmax_j; out = attn@v
// One block per (b,i); 128 threads.
// ---------------------------------------------------------------------------
__global__ __launch_bounds__(128) void k_attention(const float* __restrict__ q,
                                                   const float* __restrict__ k,
                                                   const float* __restrict__ v,
                                                   bf16_t* __restrict__ attOut) {
    __shared__ float qrow[CN];
    __shared__ float sc[FN];
    __shared__ float sred[2];
    int b = blockIdx.x / FN, i = blockIdx.x % FN;
    int tid = threadIdx.x;

    const float* qr = q + (size_t)(b * FN + i) * CN;
    for (int c = tid; c < CN; c += 128) qrow[c] = qr[c];
    __syncthreads();

    if (tid < FN) {
        const float* kr = k + (size_t)(b * FN + tid) * CN;
        float s = 0.0f;
        for (int c = 0; c < CN; ++c) s += fmaxf(qrow[c] + kr[c], 0.0f);
        sc[tid] = s;
    }
    __syncthreads();
    if (tid == 0) {
        float m = sc[0];
        for (int j = 1; j < FN; ++j) m = fmaxf(m, sc[j]);
        sred[0] = m;
    }
    __syncthreads();
    if (tid < FN) sc[tid] = expf(sc[tid] - sred[0]);
    __syncthreads();
    if (tid == 0) {
        float s = 0.0f;
        for (int j = 0; j < FN; ++j) s += sc[j];
        sred[1] = 1.0f / s;
    }
    __syncthreads();
    if (tid < FN) sc[tid] *= sred[1];
    __syncthreads();

    for (int c = tid; c < CN; c += 128) {
        const float* vb = v + (size_t)b * FN * CN + c;
        float acc = 0.0f;
        for (int j = 0; j < FN; ++j) acc += sc[j] * vb[(size_t)j * CN];
        attOut[(size_t)(b * FN + i) * CN + c] = f2bf(acc);
    }
}

// ---------------------------------------------------------------------------
// Split rows f<40 -> xa [B,40,C] ; f>=40 -> xb [B,41,C]
// ---------------------------------------------------------------------------
__global__ void k_split(const float* __restrict__ x, float* __restrict__ xa,
                        float* __restrict__ xb) {
    int idx = blockIdx.x * blockDim.x + threadIdx.x;
    if (idx >= BN * FN * CN) return;
    int c = idx % CN, row = idx / CN;
    int b = row / FN, f = row % FN;
    float val = x[idx];
    if (f < HALF_F) xa[((size_t)b * HALF_F + f) * CN + c] = val;
    else            xb[((size_t)b * F2 + (f - HALF_F)) * CN + c] = val;
}

// ---------------------------------------------------------------------------
// Forward Haar DWT: hf [B,41,512] -> ll,lh,hl,hh [B,21,256] (row 41 = zero pad)
// ---------------------------------------------------------------------------
__global__ void k_dwt_fwd(const float* __restrict__ hf, float* __restrict__ ll,
                          float* __restrict__ lh, float* __restrict__ hl,
                          float* __restrict__ hh) {
    int idx = blockIdx.x * blockDim.x + threadIdx.x;
    if (idx >= BN * HH * HW) return;
    int c = idx % HW, rem = idx / HW;
    int r = rem % HH, b = rem / HH;
    const float s = INV_SQRT2f;
    int r0 = 2 * r, r1 = 2 * r + 1;
    const float* base = hf + (size_t)b * F2 * CN;
    float e0 = base[(size_t)r0 * CN + 2 * c];
    float o0 = base[(size_t)r0 * CN + 2 * c + 1];
    float la0 = (e0 + o0) * s, ld0 = (e0 - o0) * s;
    float la1 = 0.0f, ld1 = 0.0f;
    if (r1 < F2) {
        float e1 = base[(size_t)r1 * CN + 2 * c];
        float o1 = base[(size_t)r1 * CN + 2 * c + 1];
        la1 = (e1 + o1) * s; ld1 = (e1 - o1) * s;
    }
    size_t o = (size_t)(b * HH + r) * HW + c;
    ll[o] = (la0 + la1) * s;
    lh[o] = (la0 - la1) * s;
    hl[o] = (ld0 + ld1) * s;
    hh[o] = (ld0 - ld1) * s;
}

// ---------------------------------------------------------------------------
// Bilinear upsample ll [B,21,256] -> bf16 [B,41,512]
// ---------------------------------------------------------------------------
__global__ void k_upsample(const float* __restrict__ ll, bf16_t* __restrict__ out) {
    int idx = blockIdx.x * blockDim.x + threadIdx.x;
    if (idx >= BN * F2 * CN) return;
    int x = idx % CN, rem = idx / CN;
    int y = rem % F2, b = rem / F2;
    float ys = y * (float)(HH - 1) / (float)(F2 - 1);
    float xs = x * (float)(HW - 1) / (float)(CN - 1);
    int y0 = (int)floorf(ys); int y1 = min(y0 + 1, HH - 1);
    int x0 = (int)floorf(xs); int x1 = min(x0 + 1, HW - 1);
    float wy = ys - y0, wx = xs - x0;
    const float* base = ll + (size_t)b * HH * HW;
    float t0 = base[(size_t)y0 * HW + x0] * (1.0f - wy) + base[(size_t)y1 * HW + x0] * wy;
    float t1 = base[(size_t)y0 * HW + x1] * (1.0f - wy) + base[(size_t)y1 * HW + x1] * wy;
    out[idx] = f2bf(t0 * (1.0f - wx) + t1 * wx);
}

// ---------------------------------------------------------------------------
// Bilinear downsample [B,41,512] -> [B,21,256]
// ---------------------------------------------------------------------------
__global__ void k_downsample(const float* __restrict__ in, float* __restrict__ out) {
    int idx = blockIdx.x * blockDim.x + threadIdx.x;
    if (idx >= BN * HH * HW) return;
    int x = idx % HW, rem = idx / HW;
    int y = rem % HH, b = rem / HH;
    float ys = y * (float)(F2 - 1) / (float)(HH - 1);
    float xs = x * (float)(CN - 1) / (float)(HW - 1);
    int y0 = (int)floorf(ys); int y1 = min(y0 + 1, F2 - 1);
    int x0 = (int)floorf(xs); int x1 = min(x0 + 1, CN - 1);
    float wy = ys - y0, wx = xs - x0;
    const float* base = in + (size_t)b * F2 * CN;
    float t0 = base[(size_t)y0 * CN + x0] * (1.0f - wy) + base[(size_t)y1 * CN + x0] * wy;
    float t1 = base[(size_t)y0 * CN + x1] * (1.0f - wy) + base[(size_t)y1 * CN + x1] * wy;
    out[idx] = t0 * (1.0f - wx) + t1 * wx;
}

// ---------------------------------------------------------------------------
// Inverse Haar DWT (+ residual): y2[b,f,c] = xb + idwt(low_ds, lh, hl, hh)
// ---------------------------------------------------------------------------
__global__ void k_idwt_res(const float* __restrict__ low, const float* __restrict__ lh,
                           const float* __restrict__ hl, const float* __restrict__ hh,
                           const float* __restrict__ xb, float* __restrict__ y2) {
    int idx = blockIdx.x * blockDim.x + threadIdx.x;
    if (idx >= BN * F2 * CN) return;
    int c = idx % CN, rem = idx / CN;
    int f = rem % F2, b = rem / F2;
    const float s = INV_SQRT2f;
    int hr = f >> 1; float fs = (f & 1) ? -1.0f : 1.0f;
    int wc = c >> 1; float cs = (c & 1) ? -1.0f : 1.0f;
    size_t o2 = (size_t)(b * HH + hr) * HW + wc;
    float Ah = (low[o2] + fs * lh[o2]) * s;   // idwt_h of (low, lh)
    float Dh = (hl[o2]  + fs * hh[o2]) * s;   // idwt_h of (hl, hh)
    float rec = (Ah + cs * Dh) * s;           // idwt_w
    y2[idx] = xb[idx] + rec;
}

// ---------------------------------------------------------------------------
// Concat y1 [B,40,C] and y2 [B,41,C] -> out [B,81,C]
// ---------------------------------------------------------------------------
__global__ void k_concat(const float* __restrict__ y1, const float* __restrict__ y2,
                         float* __restrict__ out) {
    int idx = blockIdx.x * blockDim.x + threadIdx.x;
    if (idx >= BN * FN * CN) return;
    int c = idx % CN, row = idx / CN;
    int b = row / FN, f = row % FN;
    out[idx] = (f < HALF_F)
             ? y1[((size_t)b * HALF_F + f) * CN + c]
             : y2[((size_t)b * F2 + (f - HALF_F)) * CN + c];
}

// ---------------------------------------------------------------------------
// Host launcher
// ---------------------------------------------------------------------------
extern "C" void kernel_launch(void* const* d_in, const int* in_sizes, int n_in,
                              void* d_out, int out_size, void* d_ws, size_t ws_size,
                              hipStream_t stream) {
    (void)in_sizes; (void)n_in; (void)out_size; (void)ws_size;

    const float* x       = (const float*)d_in[0];
    const float* norm1_w = (const float*)d_in[1];
    const float* norm1_b = (const float*)d_in[2];
    const float* wq      = (const float*)d_in[3];
    const float* wk      = (const float*)d_in[4];
    const float* wv      = (const float*)d_in[5];
    const float* wp      = (const float*)d_in[6];
    const float* bp      = (const float*)d_in[7];
    const float* norm2_w = (const float*)d_in[8];
    const float* norm2_b = (const float*)d_in[9];
    const float* w11     = (const float*)d_in[10];
    const float* b11     = (const float*)d_in[11];
    const float* w12     = (const float*)d_in[12];
    const float* b12     = (const float*)d_in[13];
    const float* norm3_w = (const float*)d_in[14];
    const float* norm3_b = (const float*)d_in[15];
    const float* w21     = (const float*)d_in[16];
    const float* b21     = (const float*)d_in[17];
    const float* w22     = (const float*)d_in[18];
    const float* b22     = (const float*)d_in[19];
    float* out = (float*)d_out;

    // ---- carve workspace --------------------------------------------------
    char* base = (char*)d_ws;
    size_t off = 0;
    auto alloc = [&](size_t bytes) -> char* {
        size_t o = (off + 255) & ~(size_t)255;
        off = o + bytes;
        return base + o;
    };
    const int M  = BN * FN;        // 2592
    const int M1 = BN * HALF_F;    // 1280
    const int M2 = BN * F2;        // 1312
    const int LL = BN * HH * HW;   // 172032

    bf16_t* h_bf   = (bf16_t*)alloc((size_t)M  * CN * 2);
    bf16_t* wqT    = (bf16_t*)alloc((size_t)CN * CN * 2);
    bf16_t* wkT    = (bf16_t*)alloc((size_t)CN * CN * 2);
    bf16_t* wvT    = (bf16_t*)alloc((size_t)CN * CN * 2);
    bf16_t* wpT    = (bf16_t*)alloc((size_t)CN * CN * 2);
    bf16_t* w11T   = (bf16_t*)alloc((size_t)HIDN * CN * 2);
    bf16_t* w12T   = (bf16_t*)alloc((size_t)CN * HIDN * 2);
    bf16_t* w21T   = (bf16_t*)alloc((size_t)HIDN * CN * 2);
    bf16_t* w22T   = (bf16_t*)alloc((size_t)CN * HIDN * 2);
    float*  qf     = (float*)alloc((size_t)M * CN * 4);
    float*  kf     = (float*)alloc((size_t)M * CN * 4);
    float*  vf     = (float*)alloc((size_t)M * CN * 4);
    bf16_t* att_bf = (bf16_t*)alloc((size_t)M * CN * 2);
    float*  xpost  = (float*)alloc((size_t)M * CN * 4);
    float*  xa     = (float*)alloc((size_t)M1 * CN * 4);
    float*  xb     = (float*)alloc((size_t)M2 * CN * 4);
    bf16_t* ha_bf  = (bf16_t*)alloc((size_t)M1 * CN * 2);
    bf16_t* t1_bf  = (bf16_t*)alloc((size_t)M1 * HIDN * 2);
    float*  y1     = (float*)alloc((size_t)M1 * CN * 4);
    float*  hf     = (float*)alloc((size_t)M2 * CN * 4);
    float*  ll     = (float*)alloc((size_t)LL * 4);
    float*  lh     = (float*)alloc((size_t)LL * 4);
    float*  hl     = (float*)alloc((size_t)LL * 4);
    float*  hh     = (float*)alloc((size_t)LL * 4);
    bf16_t* low_bf = (bf16_t*)alloc((size_t)M2 * CN * 2);
    bf16_t* t2_bf  = (bf16_t*)alloc((size_t)M2 * HIDN * 2);
    float*  lowout = (float*)alloc((size_t)M2 * CN * 4);
    float*  low_ds = (float*)alloc((size_t)LL * 4);
    float*  y2     = (float*)alloc((size_t)M2 * CN * 4);

    auto grid1d = [](int n) { return dim3((n + 255) / 256); };
    auto gemm_grid = [](int m, int n) { return dim3((m + 31) / 32, (n + 255) / 256); };

    // 1) weight prep (fp32 -> bf16 transposed)
    k_transpose_bf16<<<grid1d(CN * CN),   256, 0, stream>>>(wq,  wqT,  CN, CN);
    k_transpose_bf16<<<grid1d(CN * CN),   256, 0, stream>>>(wk,  wkT,  CN, CN);
    k_transpose_bf16<<<grid1d(CN * CN),   256, 0, stream>>>(wv,  wvT,  CN, CN);
    k_transpose_bf16<<<grid1d(CN * CN),   256, 0, stream>>>(wp,  wpT,  CN, CN);
    k_transpose_bf16<<<grid1d(CN * HIDN), 256, 0, stream>>>(w11, w11T, CN, HIDN);
    k_transpose_bf16<<<grid1d(HIDN * CN), 256, 0, stream>>>(w12, w12T, HIDN, CN);
    k_transpose_bf16<<<grid1d(CN * HIDN), 256, 0, stream>>>(w21, w21T, CN, HIDN);
    k_transpose_bf16<<<grid1d(HIDN * CN), 256, 0, stream>>>(w22, w22T, HIDN, CN);

    // 2) LN1 -> h (bf16)
    k_ln512<<<M, 256, 0, stream>>>(x, norm1_w, norm1_b, nullptr, h_bf);

    // 3) q,k,v projections (WMMA)
    k_gemm_wmma<<<gemm_grid(M, CN), 256, 0, stream>>>(h_bf, wqT, nullptr, nullptr, qf, nullptr, M, CN, CN, 0);
    k_gemm_wmma<<<gemm_grid(M, CN), 256, 0, stream>>>(h_bf, wkT, nullptr, nullptr, kf, nullptr, M, CN, CN, 0);
    k_gemm_wmma<<<gemm_grid(M, CN), 256, 0, stream>>>(h_bf, wvT, nullptr, nullptr, vf, nullptr, M, CN, CN, 0);

    // 4) additive attention + softmax + attn@v
    k_attention<<<M, 128, 0, stream>>>(qf, kf, vf, att_bf);

    // 5) output projection + residual: xpost = x + att@wp + bp  (WMMA)
    k_gemm_wmma<<<gemm_grid(M, CN), 256, 0, stream>>>(att_bf, wpT, bp, x, xpost, nullptr, M, CN, CN, 0);

    // 6) split rows
    k_split<<<grid1d(M * CN), 256, 0, stream>>>(xpost, xa, xb);

    // 7-9) branch 1 MLP: y1 = xa + gelu(ln2(xa)@w11+b11)@w12+b12  (WMMA)
    k_ln512<<<M1, 256, 0, stream>>>(xa, norm2_w, norm2_b, nullptr, ha_bf);
    k_gemm_wmma<<<gemm_grid(M1, HIDN), 256, 0, stream>>>(ha_bf, w11T, b11, nullptr, nullptr, t1_bf, M1, HIDN, CN, 1);
    k_gemm_wmma<<<gemm_grid(M1, CN),   256, 0, stream>>>(t1_bf, w12T, b12, xa, y1, nullptr, M1, CN, HIDN, 0);

    // 10-16) branch 2: LN3 -> DWT -> upsample -> MLP (WMMA) -> downsample -> IDWT
    k_ln512<<<M2, 256, 0, stream>>>(xb, norm3_w, norm3_b, hf, nullptr);
    k_dwt_fwd<<<grid1d(LL), 256, 0, stream>>>(hf, ll, lh, hl, hh);
    k_upsample<<<grid1d(M2 * CN), 256, 0, stream>>>(ll, low_bf);
    k_gemm_wmma<<<gemm_grid(M2, HIDN), 256, 0, stream>>>(low_bf, w21T, b21, nullptr, nullptr, t2_bf, M2, HIDN, CN, 1);
    k_gemm_wmma<<<gemm_grid(M2, CN),   256, 0, stream>>>(t2_bf, w22T, b22, nullptr, lowout, nullptr, M2, CN, HIDN, 0);
    k_downsample<<<grid1d(LL), 256, 0, stream>>>(lowout, low_ds);
    k_idwt_res<<<grid1d(M2 * CN), 256, 0, stream>>>(low_ds, lh, hl, hh, xb, y2);

    // 17) concat
    k_concat<<<grid1d(M * CN), 256, 0, stream>>>(y1, y2, out);
}
